// AssociativeMemoryL1_29661044146658
// MI455X (gfx1250) — matrix operand order repaired and checked by reading, not compiled, not run on hardware
//
#include <hip/hip_runtime.h>
#include <hip/hip_bf16.h>

typedef unsigned short u16_t;
typedef __attribute__((ext_vector_type(16))) __bf16 v16bf;
typedef __attribute__((ext_vector_type(8)))  float  v8f;

union Frag { uint4 u[2]; v16bf v; };

// ---------------- constants ----------------
#define NTOK   16384        // B*S
#define DMODEL 2048
#define HDK    1024         // H*DK
#define PKVQ   3072         // Wk|Wv|Wq packed rows
#define NHEAD  16
#define HS     64           // DK == DV
#define DECAYF 0.99f

// LDS row pitch (halfwords) for 32-wide K tiles: 32 + 8 pad -> 80B (16B multiple)
#define LPITCH 40
// LDS row pitch for the 128-wide transposed staging in mem_update
#define TPITCH 136

// -------- CDNA5 async-to-LDS path (guarded; falls back to sync staging) ------
#if __has_builtin(__builtin_amdgcn_global_load_async_to_lds_b128)
#define HAVE_ASYNC_LDS 1
// The builtin takes pointers to a 4 x int vector (b128 payload):
//   (v4i addrspace(1)* gsrc, v4i addrspace(3)* ldst, int offset, int cpol)
typedef int v4i_vs __attribute__((vector_size(16)));

__device__ __forceinline__ void async_cp16(const u16_t* g, u16_t* l) {
    // AS1 pointers are 64-bit; AS3 pointers are the 32-bit LDS offset
    __builtin_amdgcn_global_load_async_to_lds_b128(
        (__attribute__((address_space(1))) v4i_vs*)(unsigned long long)(size_t)g,
        (__attribute__((address_space(3))) v4i_vs*)(unsigned int)(size_t)l,
        0, 0);
}
__device__ __forceinline__ void wait_async0() {
#if __has_builtin(__builtin_amdgcn_s_wait_asynccnt)
    __builtin_amdgcn_s_wait_asynccnt(0);
#else
    asm volatile("s_wait_asynccnt 0" ::: "memory");
#endif
}
#else
#define HAVE_ASYNC_LDS 0
#endif

__device__ __forceinline__ u16_t f32_to_bf16(float f) {
    unsigned int u = __float_as_uint(f);
    unsigned int lsb = (u >> 16) & 1u;
    u += 0x7fffu + lsb;               // round-to-nearest-even
    return (u16_t)(u >> 16);
}

// ---------------- f32 -> bf16 convert ----------------
__global__ __launch_bounds__(256) void cvt_kernel(const float* __restrict__ in,
                                                  u16_t* __restrict__ out, int n) {
    int i = blockIdx.x * blockDim.x + threadIdx.x;
    int stride = gridDim.x * blockDim.x;
    for (; i < n; i += stride) out[i] = f32_to_bf16(in[i]);
}

// ---------------- MeffT[d][p] = sum_v mem[p*64+v] * Wo[d*1024 + (p>>6)*64 + v] ----------------
__global__ __launch_bounds__(256) void meff_kernel(const float* __restrict__ mem,
                                                   const float* __restrict__ Wo,
                                                   u16_t* __restrict__ MeffT) {
    int idx = blockIdx.x * blockDim.x + threadIdx.x;   // d*1024 + p
    int p = idx & (HDK - 1);
    int d = idx >> 10;
    int h = p >> 6;
    const float* mrow = mem + p * HS;
    const float* wrow = Wo + (size_t)d * HDK + h * HS;
    float s = 0.f;
#pragma unroll 8
    for (int v = 0; v < HS; ++v) s += mrow[v] * wrow[v];
    MeffT[idx] = f32_to_bf16(s);
}

// ---------------- shared GEMM building blocks ----------------
// Stage a 128-row x 32-halfword tile into LDS (pitch LPITCH), 256 threads.
__device__ __forceinline__ void stage_tile(const u16_t* __restrict__ gsrc, int ld,
                                           u16_t* __restrict__ lds, int tid) {
#pragma unroll
    for (int i = 0; i < 2; ++i) {
        int c   = tid + i * 256;      // 0..511 chunk id
        int row = c >> 2;
        int ch  = c & 3;
        uint4 val = *(const uint4*)(gsrc + (size_t)row * ld + ch * 8);
        *(uint4*)(lds + row * LPITCH + ch * 8) = val;
    }
}

#if HAVE_ASYNC_LDS
__device__ __forceinline__ void stage_tile_async(const u16_t* __restrict__ gsrc, int ld,
                                                 u16_t* __restrict__ lds, int tid) {
#pragma unroll
    for (int i = 0; i < 2; ++i) {
        int c   = tid + i * 256;
        int row = c >> 2;
        int ch  = c & 3;
        async_cp16(gsrc + (size_t)row * ld + ch * 8, lds + row * LPITCH + ch * 8);
    }
}
#endif

// One 32-deep MAC step for a 32x64 wave tile (2 x 4 accumulators of 16x16).
__device__ __forceinline__ void wave_mac(const u16_t* __restrict__ As,
                                         const u16_t* __restrict__ Bs,
                                         int wm, int wn, int lane, v8f acc[2][4]) {
    const int lhalf = lane & 15;
    const int hi    = lane >> 4;
    Frag fa[2], fb[4];
#pragma unroll
    for (int i = 0; i < 2; ++i) {
        const u16_t* ar = As + (wm + i * 16 + lhalf) * LPITCH;
        fa[i].u[0] = *(const uint4*)(ar + hi * 8);        // K {0..7}/{8..15}
        fa[i].u[1] = *(const uint4*)(ar + 16 + hi * 8);   // K {16..23}/{24..31}
    }
#pragma unroll
    for (int j = 0; j < 4; ++j) {
        const u16_t* br = Bs + (wn + j * 16 + lhalf) * LPITCH;
        fb[j].u[0] = *(const uint4*)(br + hi * 8);        // K {0..7}/{8..15}
        fb[j].u[1] = *(const uint4*)(br + 16 + hi * 8);   // K {16..23}/{24..31}
    }
#pragma unroll
    for (int i = 0; i < 2; ++i)
#pragma unroll
        for (int j = 0; j < 4; ++j)
            acc[i][j] = __builtin_amdgcn_wmma_f32_16x16x32_bf16(
                false, fa[i].v, false, fb[j].v, (short)0, acc[i][j], false, false);
}

// K-loop over one GEMM pass. Async path: double-buffered software pipeline with
// one barrier per K-step; fallback: synchronous staging with two barriers.
__device__ __forceinline__ void gemm_loop(const u16_t* __restrict__ A, int lda,
                                          const u16_t* __restrict__ B, int ldb,
                                          int Klen,
                                          u16_t* As0, u16_t* Bs0,
                                          u16_t* As1, u16_t* Bs1,
                                          int tid, int wm, int wn, int lane,
                                          v8f acc[2][4]) {
#if HAVE_ASYNC_LDS
    const int nit = Klen >> 5;
    stage_tile_async(A, lda, As0, tid);
    stage_tile_async(B, ldb, Bs0, tid);
    for (int it = 0; it < nit; ++it) {
        wait_async0();            // this wave's async LDS writes landed
        __syncthreads();          // every wave's writes landed
        if (it + 1 < nit) {       // prefetch next K tile into the other buffer
            u16_t* An = ((it + 1) & 1) ? As1 : As0;
            u16_t* Bn = ((it + 1) & 1) ? Bs1 : Bs0;
            stage_tile_async(A + (it + 1) * 32, lda, An, tid);
            stage_tile_async(B + (it + 1) * 32, ldb, Bn, tid);
        }
        wave_mac((it & 1) ? As1 : As0, (it & 1) ? Bs1 : Bs0, wm, wn, lane, acc);
    }
    __syncthreads();              // LDS reusable by a following pass
#else
    for (int k0 = 0; k0 < Klen; k0 += 32) {
        stage_tile(A + k0, lda, As0, tid);
        stage_tile(B + k0, ldb, Bs0, tid);
        if (k0 + 32 < Klen) {
            __builtin_prefetch(A + (size_t)(tid >> 1) * lda + k0 + 32, 0, 0);
            __builtin_prefetch(B + (size_t)(tid >> 1) * ldb + k0 + 32, 0, 0);
        }
        __syncthreads();
        wave_mac(As0, Bs0, wm, wn, lane, acc);
        __syncthreads();
    }
#endif
}

// ---------------- kvq projection: Out[n][p] = sum_d X[n][d] * W[p][d] ----------------
__global__ __launch_bounds__(256) void proj_gemm(const u16_t* __restrict__ X,
                                                 const u16_t* __restrict__ W,
                                                 u16_t* __restrict__ Out) {
    __shared__ u16_t As0[128 * LPITCH], Bs0[128 * LPITCH];
    __shared__ u16_t As1[128 * LPITCH], Bs1[128 * LPITCH];
    const int tid  = threadIdx.x;
    const int lane = tid & 31;
    const int wv   = tid >> 5;
    const int wm   = (wv >> 1) * 32;
    const int wn   = (wv & 1) * 64;
    const int m0   = blockIdx.y * 128;
    const int p0   = blockIdx.x * 128;

    v8f zero = {0.f, 0.f, 0.f, 0.f, 0.f, 0.f, 0.f, 0.f};
    v8f acc[2][4];
#pragma unroll
    for (int i = 0; i < 2; ++i)
#pragma unroll
        for (int j = 0; j < 4; ++j) acc[i][j] = zero;

    gemm_loop(X + (size_t)m0 * DMODEL, DMODEL,
              W + (size_t)p0 * DMODEL, DMODEL, DMODEL,
              As0, Bs0, As1, Bs1, tid, wm, wn, lane, acc);

    const int lhalf = lane & 15, hi = lane >> 4;
#pragma unroll
    for (int i = 0; i < 2; ++i)
#pragma unroll
        for (int j = 0; j < 4; ++j)
#pragma unroll
            for (int r = 0; r < 8; ++r) {
                int row = m0 + wm + i * 16 + hi * 8 + r;
                int col = p0 + wn + j * 16 + lhalf;
                Out[(size_t)row * PKVQ + col] = f32_to_bf16(acc[i][j][r]);
            }
}

// ---------------- output: out = hidden + sigmoid(X Wg^T) * (Q MeffT^T) ----------------
__global__ __launch_bounds__(256) void out_gemm(const u16_t* __restrict__ X,
                                                const u16_t* __restrict__ Wg,
                                                const u16_t* __restrict__ kqv,
                                                const u16_t* __restrict__ MeffT,
                                                const float* __restrict__ hidden,
                                                float* __restrict__ out) {
    __shared__ u16_t As0[128 * LPITCH], Bs0[128 * LPITCH];
    __shared__ u16_t As1[128 * LPITCH], Bs1[128 * LPITCH];
    const int tid  = threadIdx.x;
    const int lane = tid & 31;
    const int wv   = tid >> 5;
    const int wm   = (wv >> 1) * 32;
    const int wn   = (wv & 1) * 64;
    const int m0   = blockIdx.y * 128;
    const int p0   = blockIdx.x * 128;   // output feature tile (0..2047)

    v8f zero = {0.f, 0.f, 0.f, 0.f, 0.f, 0.f, 0.f, 0.f};
    v8f accg[2][4], accr[2][4];
#pragma unroll
    for (int i = 0; i < 2; ++i)
#pragma unroll
        for (int j = 0; j < 4; ++j) { accg[i][j] = zero; accr[i][j] = zero; }

    // pass 1: gate logits, K = 2048
    gemm_loop(X + (size_t)m0 * DMODEL, DMODEL,
              Wg + (size_t)p0 * DMODEL, DMODEL, DMODEL,
              As0, Bs0, As1, Bs1, tid, wm, wn, lane, accg);
    // pass 2: delta = q @ MeffT^T, K = 1024 (q = kqv columns 2048..3071)
    gemm_loop(kqv + 2048 + (size_t)m0 * PKVQ, PKVQ,
              MeffT + (size_t)p0 * HDK, HDK, HDK,
              As0, Bs0, As1, Bs1, tid, wm, wn, lane, accr);

    const int lhalf = lane & 15, hi = lane >> 4;
#pragma unroll
    for (int i = 0; i < 2; ++i)
#pragma unroll
        for (int j = 0; j < 4; ++j)
#pragma unroll
            for (int r = 0; r < 8; ++r) {
                int row = m0 + wm + i * 16 + hi * 8 + r;
                int col = p0 + wn + j * 16 + lhalf;
                size_t idx = (size_t)row * DMODEL + col;
                float g = 1.0f / (1.0f + expf(-accg[i][j][r]));
                out[idx] = hidden[idx] + g * accr[i][j][r];
            }
}

// ---------------- memory update: M'[h] = 0.99*M[h] + K_h^T V_h ----------------
__global__ __launch_bounds__(128) void mem_update(const u16_t* __restrict__ kqv,
                                                  const float* __restrict__ mem_in,
                                                  float* __restrict__ mem_out) {
    __shared__ u16_t kT[HS * TPITCH];   // [col a][n]
    __shared__ u16_t vT[HS * TPITCH];   // [col b][n]
    const int h    = blockIdx.x;
    const int tid  = threadIdx.x;        // 128 threads = 4 waves
    const int lane = tid & 31;
    const int wv   = tid >> 5;           // owns b-range wv*16..wv*16+15

    const u16_t* kbase = kqv + h * HS;           // k: cols 0..1023
    const u16_t* vbase = kqv + HDK + h * HS;     // v: cols 1024..2047

    v8f zero = {0.f, 0.f, 0.f, 0.f, 0.f, 0.f, 0.f, 0.f};
    v8f acc[4];
#pragma unroll
    for (int i = 0; i < 4; ++i) acc[i] = zero;

    const int lhalf = lane & 15, hi = lane >> 4;

    for (int n0 = 0; n0 < NTOK; n0 += 128) {
        // transpose-stage 128 tokens x 64 features for k and v
        const u16_t* krow = kbase + (size_t)(n0 + tid) * PKVQ;
        const u16_t* vrow = vbase + (size_t)(n0 + tid) * PKVQ;
#pragma unroll
        for (int cc = 0; cc < HS; cc += 8) {
            uint4 kvv = *(const uint4*)(krow + cc);
            uint4 vvv = *(const uint4*)(vrow + cc);
            const u16_t* kp = (const u16_t*)&kvv;
            const u16_t* vp = (const u16_t*)&vvv;
#pragma unroll
            for (int e = 0; e < 8; ++e) {
                kT[(cc + e) * TPITCH + tid] = kp[e];
                vT[(cc + e) * TPITCH + tid] = vp[e];
            }
        }
        __syncthreads();
#pragma unroll
        for (int ks = 0; ks < 4; ++ks) {
            Frag fb;
            const u16_t* br = vT + (wv * 16 + lhalf) * TPITCH + ks * 32;
            fb.u[0] = *(const uint4*)(br + hi * 8);
            fb.u[1] = *(const uint4*)(br + 16 + hi * 8);
#pragma unroll
            for (int i = 0; i < 4; ++i) {
                Frag fa;
                const u16_t* ar = kT + (i * 16 + lhalf) * TPITCH + ks * 32;
                fa.u[0] = *(const uint4*)(ar + hi * 8);
                fa.u[1] = *(const uint4*)(ar + 16 + hi * 8);
                acc[i] = __builtin_amdgcn_wmma_f32_16x16x32_bf16(
                    false, fa.v, false, fb.v, (short)0, acc[i], false, false);
            }
        }
        __syncthreads();
    }

#pragma unroll
    for (int i = 0; i < 4; ++i)
#pragma unroll
        for (int r = 0; r < 8; ++r) {
            int a = i * 16 + hi * 8 + r;
            int b = wv * 16 + lhalf;
            int idx = h * (HS * HS) + a * HS + b;
            mem_out[idx] = DECAYF * mem_in[idx] + acc[i][r];
        }
}

// ---------------- host orchestration ----------------
extern "C" void kernel_launch(void* const* d_in, const int* in_sizes, int n_in,
                              void* d_out, int out_size, void* d_ws, size_t ws_size,
                              hipStream_t stream) {
    const float* hidden = (const float*)d_in[0];   // (4,4096,2048)
    const float* memory = (const float*)d_in[1];   // (16,64,64)
    const float* Wk     = (const float*)d_in[2];   // (1024,2048)
    const float* Wv     = (const float*)d_in[3];
    const float* Wq     = (const float*)d_in[4];
    const float* Wg     = (const float*)d_in[5];   // (2048,2048)
    const float* Wo     = (const float*)d_in[6];   // (2048,1024)

    float* out     = (float*)d_out;
    float* mem_out = out + (size_t)NTOK * DMODEL;

    // workspace layout (bf16 halfwords), ~193 MB total
    u16_t* Xb    = (u16_t*)d_ws;                        // 33,554,432
    u16_t* Wkvq  = Xb + (size_t)NTOK * DMODEL;          //  6,291,456 (Wk|Wv|Wq)
    u16_t* Wgb   = Wkvq + (size_t)PKVQ * DMODEL;        //  4,194,304
    u16_t* MeffT = Wgb + (size_t)DMODEL * DMODEL;       //  2,097,152
    u16_t* kqv   = MeffT + (size_t)DMODEL * HDK;        // 50,331,648

    cvt_kernel<<<4096, 256, 0, stream>>>(hidden, Xb, NTOK * DMODEL);
    cvt_kernel<<<2048, 256, 0, stream>>>(Wk, Wkvq,               HDK * DMODEL);
    cvt_kernel<<<2048, 256, 0, stream>>>(Wv, Wkvq + (size_t)HDK * DMODEL,     HDK * DMODEL);
    cvt_kernel<<<2048, 256, 0, stream>>>(Wq, Wkvq + (size_t)2 * HDK * DMODEL, HDK * DMODEL);
    cvt_kernel<<<2048, 256, 0, stream>>>(Wg, Wgb, DMODEL * DMODEL);

    meff_kernel<<<(DMODEL * HDK) / 256, 256, 0, stream>>>(memory, Wo, MeffT);

    proj_gemm<<<dim3(PKVQ / 128, NTOK / 128), 256, 0, stream>>>(Xb, Wkvq, kqv);

    out_gemm<<<dim3(DMODEL / 128, NTOK / 128), 256, 0, stream>>>(
        Xb, Wgb, kqv, MeffT, hidden, out);

    mem_update<<<NHEAD, 128, 0, stream>>>(kqv, memory, mem_out);
}
